// EvolvingPhasorBlock_45337674776745
// MI455X (gfx1250) — matrix operand order repaired and checked by reading, not compiled, not run on hardware
//
#include <hip/hip_runtime.h>
#include <stdint.h>
#include <stddef.h>

// Problem constants (fixed by the reference setup_inputs()).
#define BB 4
#define LL 4096
#define DD 512
#define PP 128
#define VVAL 8
#define MM (BB * LL)   // 16384 rows

typedef __attribute__((ext_vector_type(16))) __bf16 v16bf;
typedef __attribute__((ext_vector_type(8)))  float  v8f;

union AFrag { v16bf v; uint32_t u[8]; };

__device__ __forceinline__ uint16_t f2bf(float f) {
    union { float f; uint32_t u; } cv; cv.f = f;
    uint32_t r = (cv.u + 0x7FFFu + ((cv.u >> 16) & 1u)) >> 16;  // RNE
    return (uint16_t)r;
}
__device__ __forceinline__ float gelu_erf(float x) {
    return 0.5f * x * (1.0f + erff(x * 0.70710678118654752440f));
}
// wave-relative LDS byte offset of a __shared__ object (addrspacecast flat->local)
__device__ __forceinline__ uint32_t lds_off(const void* p) {
    return (uint32_t)(uintptr_t)(__attribute__((address_space(3))) const void*)p;
}

// ---------------------------------------------------------------------------
// f32 -> bf16 conversion (weights)
// ---------------------------------------------------------------------------
__global__ void cvt_bf16_kernel(const float* __restrict__ src,
                                uint16_t* __restrict__ dst, int n) {
    int t = blockIdx.x * blockDim.x + threadIdx.x;
    if (t < n) dst[t] = f2bf(src[t]);
}

// ---------------------------------------------------------------------------
// prep: cumsum of x over L -> context_avg; pack x (bf16) and sk_in (bf16)
// ---------------------------------------------------------------------------
__global__ void prep_kernel(const float* __restrict__ x,
                            uint16_t* __restrict__ xb,
                            uint16_t* __restrict__ skinb) {
    int t = blockIdx.x * blockDim.x + threadIdx.x;
    if (t >= BB * DD) return;
    int b = t / DD, d = t % DD;
    const float* xp = x + (size_t)b * LL * DD + d;
    float s = 0.f;
    for (int l = 0; l < LL; ++l) {
        float v = xp[(size_t)l * DD];
        s += v;
        size_t row = (size_t)b * LL + l;
        uint16_t hv = f2bf(v);
        xb[row * DD + d] = hv;
        skinb[row * (2 * DD) + d] = hv;
        skinb[row * (2 * DD) + DD + d] = f2bf(s / (float)(l + 1));
    }
}

// ---------------------------------------------------------------------------
// Tiled bf16 WMMA GEMM: out = epilogue(A[MxK] @ W[KxN] + bias)
// 256 threads = 8 waves; 128x128 tile; K stepped by 32.
// Double-buffered LDS fed by CDNA5 async global->LDS copies (ASYNCcnt):
// copy of tile k+1 overlaps WMMA of tile k; ONE barrier per K-step.
// act: 0=none 1=gelu(erf) 2=sigmoid 3=tanh*pi
// ---------------------------------------------------------------------------
#define TM 128
#define TN 128
#define TK 32

__global__ __launch_bounds__(256)
void gemm_bf16_kernel(const uint16_t* __restrict__ A,
                      const uint16_t* __restrict__ W,
                      const float* __restrict__ bias,
                      float* __restrict__ outF,
                      uint16_t* __restrict__ outB,
                      const float* __restrict__ colscale,   // |colscale[n]| post-mult
                      const float* __restrict__ sscale,     // |*sscale| post-mult
                      const float* __restrict__ residual,   // += residual[m*N+n]
                      int M, int N, int K, int act) {
    __shared__ __align__(16) uint16_t As[2][TM * TK];  // 2 x 8KB
    __shared__ __align__(16) uint16_t Bs[2][TK * TN];  // 2 x 8KB
    const int tid = threadIdx.x;
    const int wv  = tid >> 5;
    const int ln  = tid & 31;
    const int m0  = blockIdx.y * TM;
    const int n0  = blockIdx.x * TN;

    // Per-thread staging coordinates: 16 contiguous bf16 (two b128) per tile
    // for each of A and B.
    const int f  = tid * 16;
    const int ra = f >> 5,  ca = f & 31;    // A tile: [128][32]
    const int rb = f >> 7,  cb = f & 127;   // B tile: [32][128]
    const uint32_t ldsA0 = lds_off(&As[0][f]);
    const uint32_t ldsA1 = lds_off(&As[1][f]);
    const uint32_t ldsB0 = lds_off(&Bs[0][f]);
    const uint32_t ldsB1 = lds_off(&Bs[1][f]);

    // Issue async copies of K-tile `kt` into buffer `buf`.
    // INST_OFFSET is added to BOTH the LDS and the memory address (ISA 15.18.3),
    // so one address pair covers both 16-byte halves.
    auto stage_async = [&](int kt, int buf) {
        const uint16_t* gA = A + (size_t)(m0 + ra) * K + kt * TK + ca;
        const uint16_t* gB = W + (size_t)(kt * TK + rb) * N + n0 + cb;
        uint32_t la = buf ? ldsA1 : ldsA0;
        uint32_t lb = buf ? ldsB1 : ldsB0;
        asm volatile(
            "global_load_async_to_lds_b128 %0, %2, off\n\t"
            "global_load_async_to_lds_b128 %0, %2, off offset:16\n\t"
            "global_load_async_to_lds_b128 %1, %3, off\n\t"
            "global_load_async_to_lds_b128 %1, %3, off offset:16"
            :: "v"(la), "v"(lb), "v"(gA), "v"(gB) : "memory");
    };

    v8f acc[8] = {};
    const int nk = K / TK;

    stage_async(0, 0);

    for (int kt = 0; kt < nk; ++kt) {
        const int c = kt & 1;
        // Only this buffer's copies are outstanding at this point.
        asm volatile("s_wait_asynccnt 0" ::: "memory");
        __syncthreads();                      // buf[c] complete for all waves
        if (kt + 1 < nk) stage_async(kt + 1, c ^ 1);  // overlaps WMMA below

        // A fragment (16x32 bf16): lane=M, K-halves split at lane 16 (ISA 7.12.2)
        AFrag a;
        {
            int ml  = (wv << 4) + (ln & 15);
            int khU = (ln >> 4) << 2;  // uint offset 0 or 4 (elements 0 / 8)
            const uint32_t* Ar =
                reinterpret_cast<const uint32_t*>(&As[c][ml * TK]);
#pragma unroll
            for (int j = 0; j < 4; ++j) {
                a.u[j]     = Ar[khU + j];       // K = kh + 2j, 2j+1
                a.u[4 + j] = Ar[8 + khU + j];   // K = 16 + kh + 2j
            }
        }
        // B fragments (32x16 each): lane = N column, K packed 2 per VGPR (same K order as A)
        const uint16_t* Bc = &Bs[c][ln & 15];
        const int khB = (ln >> 4) << 3;
#pragma unroll
        for (int ch = 0; ch < 8; ++ch) {
            AFrag bfr;
#pragma unroll
            for (int j = 0; j < 8; ++j) {
                int k = ((j & 4) << 2) + khB + ((j & 3) << 1);
                bfr.u[j] = (uint32_t)Bc[k * TN + ch * 16] | ((uint32_t)Bc[(k + 1) * TN + ch * 16] << 16);
            }
            acc[ch] = __builtin_amdgcn_wmma_f32_16x16x32_bf16(
                false, a.v, false, bfr.v, (short)0, acc[ch], false, false);
        }
    }

    // Epilogue: C layout lane=N, VGPR r = row (rows 8..15 in lanes 16..31)
    const int rowoff = (ln < 16) ? 0 : 8;
    const int col    = ln & 15;
    const float sc   = sscale ? fabsf(sscale[0]) : 1.0f;
#pragma unroll
    for (int ch = 0; ch < 8; ++ch) {
        int n = n0 + (ch << 4) + col;
        float bb = bias ? bias[n] : 0.0f;
        float cs = colscale ? fabsf(colscale[n]) : 1.0f;
#pragma unroll
        for (int r = 0; r < 8; ++r) {
            int m = m0 + (wv << 4) + rowoff + r;
            float y = acc[ch][r] + bb;
            if (act == 1)      y = gelu_erf(y);
            else if (act == 2) y = 1.0f / (1.0f + expf(-y));
            else if (act == 3) y = tanhf(y) * 3.14159265358979323846f;
            y = y * cs * sc;
            size_t o = (size_t)m * N + n;
            if (residual) y += residual[o];
            if (outF) outF[o] = y;
            if (outB) outB[o] = f2bf(y);
        }
    }
}

// ---------------------------------------------------------------------------
// values (V=8) + store_gate: tiny N, plain f32 dot products
// ---------------------------------------------------------------------------
__global__ void venc_kernel(const float* __restrict__ x,
                            const float* __restrict__ Wv, const float* __restrict__ bv,
                            const float* __restrict__ Wg, const float* __restrict__ bg,
                            float* __restrict__ vals, float* __restrict__ sg) {
    int t = blockIdx.x * blockDim.x + threadIdx.x;
    if (t >= MM * 9) return;
    int m = t / 9, j = t % 9;
    const float* xr = x + (size_t)m * DD;
    float s = 0.f;
    if (j < 8) {
        for (int d = 0; d < DD; ++d) s += xr[d] * Wv[d * 8 + j];
        vals[(size_t)m * 8 + j] = s + bv[j];
    } else {
        for (int d = 0; d < DD; ++d) s += xr[d] * Wg[d];
        sg[m] = 1.0f / (1.0f + expf(-(s + bg[0])));
    }
}

// ---------------------------------------------------------------------------
// mem1: serial scans over L per (b,d): phi, phasor memory, retrieval
// ---------------------------------------------------------------------------
__global__ void mem1_kernel(const float* __restrict__ omg,
                            const float* __restrict__ phini,
                            const float* __restrict__ v1,
                            const float* __restrict__ mag,
                            const float* __restrict__ qoff,
                            float* __restrict__ cosb, float* __restrict__ sinb,
                            uint16_t* __restrict__ posretb) {
    int t = blockIdx.x * blockDim.x + threadIdx.x;
    if (t >= BB * DD) return;
    int b = t / DD, d = t % DD;
    float pa = 0.f, cm = 0.f, cc = 0.f, cs = 0.f;
    const float invsD = rsqrtf((float)DD);
    for (int l = 0; l < LL; ++l) {
        size_t idx = ((size_t)b * LL + l) * DD + d;
        pa += omg[idx];
        float phi = phini[idx] + pa;
        float c, s; sincosf(phi, &s, &c);
        cosb[idx] = c; sinb[idx] = s;
        float m  = mag[idx];
        float wv = m * v1[idx];
        cm += m; cc += wv * c; cs += wv * s;
        float inv = rsqrtf(cm + 1e-8f);
        float cq, sq; sincosf(phi + qoff[idx], &sq, &cq);
        posretb[idx] = f2bf((cc * cq + cs * sq) * inv * invsD);
    }
}

// ---------------------------------------------------------------------------
// kv phasor memory: one 128-thread block (one lane per p) per batch.
// Per-step reduce over p via wave32 shuffles + LDS combine of the 4 waves.
// ---------------------------------------------------------------------------
__global__ __launch_bounds__(128)
void kv_kernel(const float* __restrict__ sphase, const float* __restrict__ qphase,
               const float* __restrict__ vals, const float* __restrict__ sg,
               float* __restrict__ kvret) {
    int b = blockIdx.x;
    int p = threadIdx.x;
    __shared__ float red[4][VVAL];
    float kvc[VVAL] = {}, kvs[VVAL] = {};
    float gsum = 0.f;
    const float invsP = rsqrtf((float)PP);
    for (int l = 0; l < LL; ++l) {
        size_t row = (size_t)b * LL + l;
        float sp = sphase[row * PP + p];
        float qp = qphase[row * PP + p];
        float g  = sg[row];
        gsum += g;
        float inv = rsqrtf(fmaxf(gsum, 1.0f));
        float scs, sss; sincosf(sp, &sss, &scs);
        float qcs, qss; sincosf(qp, &qss, &qcs);
        float part[VVAL];
#pragma unroll
        for (int v = 0; v < VVAL; ++v) {
            float gv = vals[row * VVAL + v] * g;
            kvc[v] += scs * gv;
            kvs[v] += sss * gv;
            part[v] = qcs * kvc[v] + qss * kvs[v];
        }
#pragma unroll
        for (int v = 0; v < VVAL; ++v) {
            part[v] += __shfl_xor(part[v], 16, 32);
            part[v] += __shfl_xor(part[v],  8, 32);
            part[v] += __shfl_xor(part[v],  4, 32);
            part[v] += __shfl_xor(part[v],  2, 32);
            part[v] += __shfl_xor(part[v],  1, 32);
        }
        if ((p & 31) == 0) {
#pragma unroll
            for (int v = 0; v < VVAL; ++v) red[p >> 5][v] = part[v];
        }
        __syncthreads();
        if (p < VVAL) {
            float s = red[0][p] + red[1][p] + red[2][p] + red[3][p];
            kvret[row * VVAL + p] = s * inv * invsP;
        }
        __syncthreads();
    }
}

// ---------------------------------------------------------------------------
// kv_out = kvret[M,8] @ W_kvo[8,512] + b_kvo  (K=8 -> plain f32)
// ---------------------------------------------------------------------------
__global__ void kvout_kernel(const float* __restrict__ kvret,
                             const float* __restrict__ Wkvo,
                             const float* __restrict__ bkvo,
                             float* __restrict__ out) {
    int t = blockIdx.x * blockDim.x + threadIdx.x;
    if (t >= MM * DD) return;
    int m = t / DD, d = t % DD;
    float s = bkvo[d];
#pragma unroll
    for (int v = 0; v < VVAL; ++v)
        s += kvret[(size_t)m * VVAL + v] * Wkvo[v * DD + d];
    out[t] = s;
}

// ---------------------------------------------------------------------------
// combined = [pos_out, kv_out, x*cos, x*sin] (4D=2048) -> LayerNorm -> bf16
// ---------------------------------------------------------------------------
__global__ __launch_bounds__(256)
void ln_kernel(const float* __restrict__ posout, const float* __restrict__ kvout,
               const float* __restrict__ x, const float* __restrict__ cosb,
               const float* __restrict__ sinb, const float* __restrict__ g,
               const float* __restrict__ be, uint16_t* __restrict__ outb) {
    int m = blockIdx.x;
    int t = threadIdx.x;
    __shared__ float s1[256], s2[256];
    float vbuf[8];
    float sum = 0.f, sq = 0.f;
#pragma unroll
    for (int i = 0; i < 8; ++i) {
        int j = t + i * 256;          // 0..2047
        int seg = j >> 9, dd = j & (DD - 1);
        size_t idx = (size_t)m * DD + dd;
        float v;
        if (seg == 0)      v = posout[idx];
        else if (seg == 1) v = kvout[idx];
        else if (seg == 2) v = x[idx] * cosb[idx];
        else               v = x[idx] * sinb[idx];
        vbuf[i] = v; sum += v; sq += v * v;
    }
    s1[t] = sum; s2[t] = sq;
    __syncthreads();
    for (int o = 128; o > 0; o >>= 1) {
        if (t < o) { s1[t] += s1[t + o]; s2[t] += s2[t + o]; }
        __syncthreads();
    }
    float mean = s1[0] * (1.0f / 2048.0f);
    float var  = s2[0] * (1.0f / 2048.0f) - mean * mean;
    float rinv = rsqrtf(var + 1e-5f);
#pragma unroll
    for (int i = 0; i < 8; ++i) {
        int j = t + i * 256;
        float y = (vbuf[i] - mean) * rinv * g[j] + be[j];
        outb[(size_t)m * 2048 + j] = f2bf(y);
    }
}

// ---------------------------------------------------------------------------
// host-side orchestration
// ---------------------------------------------------------------------------
extern "C" void kernel_launch(void* const* d_in, const int* in_sizes, int n_in,
                              void* d_out, int out_size, void* d_ws, size_t ws_size,
                              hipStream_t stream) {
    const float* x          = (const float*)d_in[0];
    const float* W_omega    = (const float*)d_in[1];
    const float* b_omega    = (const float*)d_in[2];
    const float* omega_sc   = (const float*)d_in[3];
    const float* W_pi1      = (const float*)d_in[4];
    const float* b_pi1      = (const float*)d_in[5];
    const float* W_pi2      = (const float*)d_in[6];
    const float* b_pi2      = (const float*)d_in[7];
    const float* W_m1v      = (const float*)d_in[8];
    const float* b_m1v      = (const float*)d_in[9];
    const float* W_m1o      = (const float*)d_in[10];
    const float* b_m1o      = (const float*)d_in[11];
    const float* W_mag      = (const float*)d_in[12];
    const float* b_mag      = (const float*)d_in[13];
    const float* mag_scale  = (const float*)d_in[14];
    const float* W_qoff     = (const float*)d_in[15];
    const float* b_qoff     = (const float*)d_in[16];
    const float* W_kenc     = (const float*)d_in[17];
    const float* b_kenc     = (const float*)d_in[18];
    const float* W_venc     = (const float*)d_in[19];
    const float* b_venc     = (const float*)d_in[20];
    const float* W_sk1      = (const float*)d_in[21];
    const float* b_sk1      = (const float*)d_in[22];
    const float* W_sk2      = (const float*)d_in[23];
    const float* b_sk2      = (const float*)d_in[24];
    const float* W_sg       = (const float*)d_in[25];
    const float* b_sg       = (const float*)d_in[26];
    const float* W_kvo      = (const float*)d_in[27];
    const float* b_kvo      = (const float*)d_in[28];
    const float* ln_g       = (const float*)d_in[29];
    const float* ln_b       = (const float*)d_in[30];
    const float* W_o1       = (const float*)d_in[31];
    const float* b_o1       = (const float*)d_in[32];
    const float* W_o2       = (const float*)d_in[33];
    const float* b_o2       = (const float*)d_in[34];
    (void)in_sizes; (void)n_in; (void)out_size; (void)ws_size;

    char* base = (char*)d_ws;
    size_t off = 0;
    auto alloc = [&](size_t bytes) -> void* {
        void* p = base + off;
        off = (off + bytes + 255) & ~(size_t)255;
        return p;
    };

    // bf16 weight copies
    uint16_t* Womb  = (uint16_t*)alloc((size_t)DD * DD * 2);
    uint16_t* Wpi1b = (uint16_t*)alloc((size_t)DD * DD * 2);
    uint16_t* Wpi2b = (uint16_t*)alloc((size_t)DD * DD * 2);
    uint16_t* Wm1vb = (uint16_t*)alloc((size_t)DD * DD * 2);
    uint16_t* Wmagb = (uint16_t*)alloc((size_t)DD * DD * 2);
    uint16_t* Wqofb = (uint16_t*)alloc((size_t)DD * DD * 2);
    uint16_t* Wm1ob = (uint16_t*)alloc((size_t)DD * DD * 2);
    uint16_t* Wkenb = (uint16_t*)alloc((size_t)DD * PP * 2);
    uint16_t* Wsk1b = (uint16_t*)alloc((size_t)2 * DD * DD * 2);
    uint16_t* Wsk2b = (uint16_t*)alloc((size_t)DD * PP * 2);
    uint16_t* Wo1b  = (uint16_t*)alloc((size_t)4 * DD * 2 * DD * 2);
    uint16_t* Wo2b  = (uint16_t*)alloc((size_t)2 * DD * DD * 2);

    // activations / scratch
    uint16_t* xb      = (uint16_t*)alloc((size_t)MM * DD * 2);
    uint16_t* skinb   = (uint16_t*)alloc((size_t)MM * 2 * DD * 2);
    float*    omega_s = (float*)alloc((size_t)MM * DD * 4);
    uint16_t* pi1b    = (uint16_t*)alloc((size_t)MM * DD * 2);
    float*    phini   = (float*)alloc((size_t)MM * DD * 4);
    float*    v1      = (float*)alloc((size_t)MM * DD * 4);
    float*    magb    = (float*)alloc((size_t)MM * DD * 4);
    float*    qoffb   = (float*)alloc((size_t)MM * DD * 4);
    float*    qphase  = (float*)alloc((size_t)MM * PP * 4);
    uint16_t* sk1b    = (uint16_t*)alloc((size_t)MM * DD * 2);
    float*    sphase  = (float*)alloc((size_t)MM * PP * 4);
    float*    vals    = (float*)alloc((size_t)MM * VVAL * 4);
    float*    sgb     = (float*)alloc((size_t)MM * 4);
    float*    cosb    = (float*)alloc((size_t)MM * DD * 4);
    float*    sinb    = (float*)alloc((size_t)MM * DD * 4);
    uint16_t* posretb = (uint16_t*)alloc((size_t)MM * DD * 2);
    float*    posout  = (float*)alloc((size_t)MM * DD * 4);
    float*    kvret   = (float*)alloc((size_t)MM * VVAL * 4);
    float*    kvout   = (float*)alloc((size_t)MM * DD * 4);
    uint16_t* cmbb    = (uint16_t*)alloc((size_t)MM * 4 * DD * 2);
    uint16_t* h1b     = (uint16_t*)alloc((size_t)MM * 2 * DD * 2);

    auto cvt = [&](const float* s, uint16_t* d, int n) {
        cvt_bf16_kernel<<<dim3((n + 255) / 256), dim3(256), 0, stream>>>(s, d, n);
    };
    cvt(W_omega, Womb,  DD * DD);   cvt(W_pi1, Wpi1b, DD * DD);
    cvt(W_pi2,  Wpi2b, DD * DD);    cvt(W_m1v, Wm1vb, DD * DD);
    cvt(W_mag,  Wmagb, DD * DD);    cvt(W_qoff, Wqofb, DD * DD);
    cvt(W_m1o,  Wm1ob, DD * DD);    cvt(W_kenc, Wkenb, DD * PP);
    cvt(W_sk1,  Wsk1b, 2 * DD * DD); cvt(W_sk2, Wsk2b, DD * PP);
    cvt(W_o1,   Wo1b,  4 * DD * 2 * DD); cvt(W_o2, Wo2b, 2 * DD * DD);

    prep_kernel<<<dim3((BB * DD + 255) / 256), dim3(256), 0, stream>>>(x, xb, skinb);

    auto gemm = [&](const uint16_t* A, const uint16_t* W, const float* bias,
                    float* oF, uint16_t* oB, const float* cscale, const float* ss,
                    const float* res, int M, int N, int K, int act) {
        gemm_bf16_kernel<<<dim3(N / TN, M / TM), dim3(256), 0, stream>>>(
            A, W, bias, oF, oB, cscale, ss, res, M, N, K, act);
    };

    // projections from x (and sk_in)
    gemm(xb,    Womb,  b_omega, omega_s, nullptr, omega_sc, nullptr, nullptr, MM, DD, DD, 0);
    gemm(xb,    Wpi1b, b_pi1,   nullptr, pi1b,    nullptr, nullptr, nullptr, MM, DD, DD, 1);
    gemm(pi1b,  Wpi2b, b_pi2,   phini,   nullptr, nullptr, nullptr, nullptr, MM, DD, DD, 0);
    gemm(xb,    Wm1vb, b_m1v,   v1,      nullptr, nullptr, nullptr, nullptr, MM, DD, DD, 0);
    gemm(xb,    Wmagb, b_mag,   magb,    nullptr, nullptr, mag_scale, nullptr, MM, DD, DD, 2);
    gemm(xb,    Wqofb, b_qoff,  qoffb,   nullptr, nullptr, nullptr, nullptr, MM, DD, DD, 0);
    gemm(xb,    Wkenb, b_kenc,  qphase,  nullptr, nullptr, nullptr, nullptr, MM, PP, DD, 3);
    gemm(skinb, Wsk1b, b_sk1,   nullptr, sk1b,    nullptr, nullptr, nullptr, MM, DD, 2 * DD, 1);
    gemm(sk1b,  Wsk2b, b_sk2,   sphase,  nullptr, nullptr, nullptr, nullptr, MM, PP, DD, 3);

    venc_kernel<<<dim3((MM * 9 + 255) / 256), dim3(256), 0, stream>>>(
        x, W_venc, b_venc, W_sg, b_sg, vals, sgb);

    mem1_kernel<<<dim3((BB * DD + 255) / 256), dim3(256), 0, stream>>>(
        omega_s, phini, v1, magb, qoffb, cosb, sinb, posretb);

    gemm(posretb, Wm1ob, b_m1o, posout, nullptr, nullptr, nullptr, nullptr, MM, DD, DD, 0);

    kv_kernel<<<dim3(BB), dim3(128), 0, stream>>>(sphase, qphase, vals, sgb, kvret);
    kvout_kernel<<<dim3((MM * DD + 255) / 256), dim3(256), 0, stream>>>(
        kvret, W_kvo, b_kvo, kvout);

    ln_kernel<<<dim3(MM), dim3(256), 0, stream>>>(
        posout, kvout, x, cosb, sinb, ln_g, ln_b, cmbb);

    gemm(cmbb, Wo1b, b_o1, nullptr, h1b, nullptr, nullptr, nullptr, MM, 2 * DD, 4 * DD, 1);
    gemm(h1b,  Wo2b, b_o2, (float*)d_out, nullptr, nullptr, nullptr, x, MM, DD, 2 * DD, 0);
}
